// GatingNetwork_67439576482233
// MI455X (gfx1250) — matrix operand order repaired and checked by reading, not compiled, not run on hardware
//
#include <hip/hip_runtime.h>
#include <math.h>

typedef __attribute__((ext_vector_type(16))) __bf16          v16bf;
typedef __attribute__((ext_vector_type(8)))  float           v8f;

#define C_DIM   2048
#define E_DIM   64
#define KSTEP   32
#define NKS     (C_DIM / KSTEP)        // 64 K-steps
#define NTILE   (E_DIM / 16)           // 4 expert tiles
#define ROWS_PER_WAVE   16
#define WAVES_PER_BLOCK 4
#define ROWS_PER_BLOCK  (ROWS_PER_WAVE * WAVES_PER_BLOCK)   // 64

// ---------------------------------------------------------------------------
// Kernel 1: per-expert-column inverse L2 norms of sim [C, E]
// ---------------------------------------------------------------------------
__global__ void colnorm_kernel(const float* __restrict__ sim, float* __restrict__ invc) {
    __shared__ float red[256];
    int e = blockIdx.x;
    float s = 0.f;
    for (int k = threadIdx.x; k < C_DIM; k += 256) {
        float v = sim[(size_t)k * E_DIM + e];
        s += v * v;
    }
    red[threadIdx.x] = s;
    __syncthreads();
    for (int off = 128; off > 0; off >>= 1) {
        if ((int)threadIdx.x < off) red[threadIdx.x] += red[threadIdx.x + off];
        __syncthreads();
    }
    if (threadIdx.x == 0) invc[e] = 1.0f / fmaxf(sqrtf(red[0]), 1e-12f);
}

// ---------------------------------------------------------------------------
// Kernel 2: repack normalized sim into per-lane WMMA B fragments (bf16).
// B fragment (32x16 KxN bf16): lane L -> column n = L&15; element j -> K = 8*(L>>4) + j (j<8), 16 + 8*(L>>4) + (j-8) (j>=8).
// Fragment f = ks*NTILE + t, stored as 16 contiguous bf16 per lane.
// ---------------------------------------------------------------------------
__global__ void pack_b_kernel(const float* __restrict__ sim, const float* __restrict__ invc,
                              __bf16* __restrict__ bpack) {
    int i    = blockIdx.x * blockDim.x + threadIdx.x;   // 0 .. NKS*NTILE*32*16-1
    int elem = i & 15;
    int lane = (i >> 4) & 31;
    int t    = (i >> 9) & 3;
    int ks   = i >> 11;
    int hi   = lane >> 4;
    int n    = lane & 15;
    int K    = ks * KSTEP + (elem < 8 ? hi * 8 + elem : 16 + hi * 8 + (elem - 8));
    int e    = t * 16 + n;
    float v  = sim[(size_t)K * E_DIM + e] * invc[e];
    bpack[(size_t)i] = (__bf16)v;
}

// ---------------------------------------------------------------------------
// Kernel 3: main GEMM.  logits[row][e] = (1/||x_row||) * sum_k x[row][k]*Bn[k][e]
// One wave handles 16 rows x 64 experts via 4 v_wmma_f32_16x16x32_bf16 accumulators.
// ---------------------------------------------------------------------------
__global__ void __launch_bounds__(128)
gemm_kernel(const float* __restrict__ x, const __bf16* __restrict__ bpack,
            float* __restrict__ logits) {
    const int wave = threadIdx.x >> 5;
    const int lane = threadIdx.x & 31;
    const int hi   = lane >> 4;                 // half-wave select
    const int r0   = blockIdx.x * ROWS_PER_BLOCK + wave * ROWS_PER_WAVE;
    const int row  = r0 + (lane & 15);
    const float* xrow = x + (size_t)row * C_DIM;
    const uint4* bq   = (const uint4*)bpack;    // 2 uint4 per 16-bf16 fragment-lane

    v8f acc0 = {}, acc1 = {}, acc2 = {}, acc3 = {};
    float ss = 0.f;

    union BFrag { uint4 q[2]; v16bf h; };

    #pragma unroll 2
    for (int ks = 0; ks < NKS; ++ks) {
        // A layout (16-bit A 16x32): lanes<16 hold K [k..k+7] and [k+16..k+23];
        // lanes>=16 hold K [k+8..k+15] and [k+24..k+31]  (k = ks*32)
        const int k = ks * KSTEP + hi * 8;
        float4 p0 = *(const float4*)(xrow + k);
        float4 p1 = *(const float4*)(xrow + k + 4);
        float4 p2 = *(const float4*)(xrow + k + 16);
        float4 p3 = *(const float4*)(xrow + k + 20);
        if (ks + 2 < NKS) __builtin_prefetch(xrow + k + 2 * KSTEP, 0, 1);

        ss += p0.x*p0.x + p0.y*p0.y + p0.z*p0.z + p0.w*p0.w;
        ss += p1.x*p1.x + p1.y*p1.y + p1.z*p1.z + p1.w*p1.w;
        ss += p2.x*p2.x + p2.y*p2.y + p2.z*p2.z + p2.w*p2.w;
        ss += p3.x*p3.x + p3.y*p3.y + p3.z*p3.z + p3.w*p3.w;

        // fp32 -> bf16 via native fptrunc; backend packs with v_cvt_pk_bf16_f32
        v16bf a;
        a[0]  = (__bf16)p0.x; a[1]  = (__bf16)p0.y; a[2]  = (__bf16)p0.z; a[3]  = (__bf16)p0.w;
        a[4]  = (__bf16)p1.x; a[5]  = (__bf16)p1.y; a[6]  = (__bf16)p1.z; a[7]  = (__bf16)p1.w;
        a[8]  = (__bf16)p2.x; a[9]  = (__bf16)p2.y; a[10] = (__bf16)p2.z; a[11] = (__bf16)p2.w;
        a[12] = (__bf16)p3.x; a[13] = (__bf16)p3.y; a[14] = (__bf16)p3.z; a[15] = (__bf16)p3.w;

        BFrag b0, b1, b2, b3;
        size_t f0 = ((size_t)(ks * NTILE + 0) * 32 + lane) * 2;
        size_t f1 = ((size_t)(ks * NTILE + 1) * 32 + lane) * 2;
        size_t f2 = ((size_t)(ks * NTILE + 2) * 32 + lane) * 2;
        size_t f3 = ((size_t)(ks * NTILE + 3) * 32 + lane) * 2;
        b0.q[0] = bq[f0]; b0.q[1] = bq[f0 + 1];
        b1.q[0] = bq[f1]; b1.q[1] = bq[f1 + 1];
        b2.q[0] = bq[f2]; b2.q[1] = bq[f2 + 1];
        b3.q[0] = bq[f3]; b3.q[1] = bq[f3 + 1];

        acc0 = __builtin_amdgcn_wmma_f32_16x16x32_bf16(false, a, false, b0.h, (short)0, acc0, false, false);
        acc1 = __builtin_amdgcn_wmma_f32_16x16x32_bf16(false, a, false, b1.h, (short)0, acc1, false, false);
        acc2 = __builtin_amdgcn_wmma_f32_16x16x32_bf16(false, a, false, b2.h, (short)0, acc2, false, false);
        acc3 = __builtin_amdgcn_wmma_f32_16x16x32_bf16(false, a, false, b3.h, (short)0, acc3, false, false);
    }

    // Finish row norms: lanes L and L+16 hold disjoint K-halves of row (L&15).
    float ssf = ss + __shfl_xor(ss, 16);
    float inv = 1.0f / fmaxf(sqrtf(ssf), 1e-12f);

    // D layout: acc[t] component r -> row M = r + 8*hi, col = t*16 + (lane&15)
    #pragma unroll
    for (int r = 0; r < 8; ++r) {
        int m = r + hi * 8;
        float invr = __shfl(inv, m);            // lane m (<16) holds row m's inv norm
        float* o = logits + (size_t)(r0 + m) * E_DIM + (lane & 15);
        o[0]  = acc0[r] * invr;
        o[16] = acc1[r] * invr;
        o[32] = acc2[r] * invr;
        o[48] = acc3[r] * invr;
    }
}

// ---------------------------------------------------------------------------
// Kernel 4: mask = (logits > gates), with top-2 fallback for inactive rows.
// One row per wave32; each lane owns experts {lane, lane+32}.
// ---------------------------------------------------------------------------
__global__ void mask_kernel(const float* __restrict__ logits, const float* __restrict__ gates,
                            float* __restrict__ mask, int nrows) {
    int lane = threadIdx.x & 31;
    int row  = blockIdx.x * (blockDim.x >> 5) + (threadIdx.x >> 5);
    if (row >= nrows) return;
    const float* lr = logits + (size_t)row * E_DIM;
    float l0 = lr[lane],        l1 = lr[lane + 32];
    float g0 = gates[lane],     g1 = gates[lane + 32];
    bool  a0 = l0 > g0,         a1 = l1 > g1;
    unsigned long long act = __ballot(a0) | __ballot(a1);
    float m0 = a0 ? 1.f : 0.f,  m1 = a1 ? 1.f : 0.f;

    if (act == 0ull) {
        // ---- top-1 ----
        float v = l0; int vi = lane;
        if (l1 > v) { v = l1; vi = lane + 32; }       // strict > keeps lower index on tie
        for (int off = 16; off > 0; off >>= 1) {
            float ov = __shfl_xor(v, off);
            int   oi = __shfl_xor(vi, off);
            if (ov > v || (ov == v && oi < vi)) { v = ov; vi = oi; }
        }
        int i1 = vi;
        // ---- top-2 (exclude i1) ----
        float c0 = (lane == i1)      ? -INFINITY : l0;
        float c1 = (lane + 32 == i1) ? -INFINITY : l1;
        float v2 = c0; int vi2 = lane;
        if (c1 > v2) { v2 = c1; vi2 = lane + 32; }
        for (int off = 16; off > 0; off >>= 1) {
            float ov = __shfl_xor(v2, off);
            int   oi = __shfl_xor(vi2, off);
            if (ov > v2 || (ov == v2 && oi < vi2)) { v2 = ov; vi2 = oi; }
        }
        int i2 = vi2;
        m0 = (lane == i1      || lane == i2)      ? 1.f : 0.f;
        m1 = (lane + 32 == i1 || lane + 32 == i2) ? 1.f : 0.f;
    }
    float* mr = mask + (size_t)row * E_DIM;
    mr[lane]      = m0;
    mr[lane + 32] = m1;
}

// ---------------------------------------------------------------------------
extern "C" void kernel_launch(void* const* d_in, const int* in_sizes, int n_in,
                              void* d_out, int out_size, void* d_ws, size_t ws_size,
                              hipStream_t stream) {
    const float* x     = (const float*)d_in[0];   // [B,T,C] fp32
    const float* sim   = (const float*)d_in[1];   // [C,E]   fp32
    const float* gates = (const float*)d_in[2];   // [E]     fp32
    // d_in[3] = temperature: mathematically irrelevant to both outputs
    // (mask = logits > gates since sigmoid(t) > 0; returned logits are pre-scale)

    const int nrows = in_sizes[0] / C_DIM;        // B*T = 16384
    float* mask   = (float*)d_out;                         // [N, E]
    float* logits = (float*)d_out + (size_t)nrows * E_DIM; // [N, E]

    float*   invc  = (float*)d_ws;                           // 64 floats
    __bf16*  bpack = (__bf16*)((char*)d_ws + 256);           // 256 KB packed B

    hipLaunchKernelGGL(colnorm_kernel, dim3(E_DIM), dim3(256), 0, stream, sim, invc);
    hipLaunchKernelGGL(pack_b_kernel, dim3((NKS * NTILE * 32 * 16) / 256), dim3(256), 0, stream,
                       sim, invc, bpack);
    hipLaunchKernelGGL(gemm_kernel, dim3(nrows / ROWS_PER_BLOCK), dim3(128), 0, stream,
                       x, bpack, logits);
    hipLaunchKernelGGL(mask_kernel, dim3(nrows / (256 / 32)), dim3(256), 0, stream,
                       logits, gates, mask, nrows);
}
